// RESIDUAL_LSTM_69054484185308
// MI455X (gfx1250) — compile-verified, matrix operand order checked by reading, not run on hardware
//
#include <hip/hip_runtime.h>
#include <hip/hip_bf16.h>

// ---------------------------------------------------------------------------
// Residual 2-layer LSTM scan for MI455X (gfx1250, wave32, WMMA bf16)
// B=512, T=512, F=128, H=512
// 32 WGs x 512 threads (16 wave32 = 4 waves/SIMD).
// B-fragment stream: 3-deep register pipeline; A fragments: 1 K-tile ahead.
// ---------------------------------------------------------------------------

#define B_SZ 512
#define T_SZ 512
#define F_SZ 128
#define H_SZ 512
#define G1   (4 * H_SZ)      // 2048 gate cols, lstm1
#define G2   (4 * F_SZ)      // 512  gate cols, lstm2
#define K1   (F_SZ + H_SZ)   // 640  ([x | h1])
#define K2   (H_SZ + F_SZ)   // 640  ([c1 | h2])
#define KT1  (K1 / 32)       // 20 K-tiles
#define KT2  (K2 / 32)       // 20 K-tiles
#define NT1  (G1 / 16)       // 128 N-tiles
#define NT2  (G2 / 16)       // 32  N-tiles
#define MT   16              // batch rows per workgroup

#define LDH  520             // bf16 row stride, h1 buffer   (bank-safe, 16B-mult)
#define LDXB 136             // bf16 row stride, x buffer
#define LDA2 648             // bf16 row stride, [c1|h2] buffer
#define LDX  132             // f32  row stride, x residual buffer

typedef __bf16 bf16;
typedef __attribute__((ext_vector_type(16))) __bf16 v16bf;
typedef __attribute__((ext_vector_type(8)))  __bf16 v8bf;
typedef __attribute__((ext_vector_type(8)))  float   v8f;

union BfBits { unsigned short u; bf16 b; };

__device__ __forceinline__ bf16 f2bf(float f) {
  union { float f; unsigned u; } x; x.f = f;
  unsigned r = (x.u + 0x7FFFu + ((x.u >> 16) & 1u)) >> 16;  // RNE
  BfBits o; o.u = (unsigned short)r; return o.b;
}

__device__ __forceinline__ float sigm(float x) {
  return 1.0f / (1.0f + __expf(-x));
}
__device__ __forceinline__ float tanh_fast(float x) {
  float e = __expf(-2.0f * fabsf(x));
  float r = (1.0f - e) / (1.0f + e);
  return copysignf(r, x);
}

// (k within 32, n within 16) -> (lane, 16-bf16 slot) per the CDNA5 ISA
// 16-bit fragment layout (lanes 0-15: K 0..7,16..23 ; lanes 16-31: K 8..15,24..31)
__device__ __forceinline__ void frag_pos(int kk, int nn, int& lane, int& slot) {
  if      (kk <  8) { lane = nn;      slot = kk;      }
  else if (kk < 16) { lane = nn + 16; slot = kk - 8;  }
  else if (kk < 24) { lane = nn;      slot = kk - 8;  }
  else              { lane = nn + 16; slot = kk - 16; }
}

// ---------------------------------------------------------------------------
// Prep: pack weights (f32 -> bf16) into exact per-lane WMMA-B fragment order,
// fold biases. Packed as [ntile][ktile][lane(32)][slot(16)] bf16; L2-resident.
//   B1[k][n] = k<F ? Wih1[n][k] : Whh1[n][k-F]      (640 x 2048)
//   B2[k][n] = k<H ? Wih2[n][k] : Whh2[n][k-H]      (640 x 512)
// ---------------------------------------------------------------------------
__global__ void lstm_pack_kernel(const float* __restrict__ Wih1, const float* __restrict__ Whh1,
                                 const float* __restrict__ bih1, const float* __restrict__ bhh1,
                                 const float* __restrict__ Wih2, const float* __restrict__ Whh2,
                                 const float* __restrict__ bih2, const float* __restrict__ bhh2,
                                 bf16* __restrict__ w1p, bf16* __restrict__ w2p,
                                 float* __restrict__ b1, float* __restrict__ b2) {
  const long total1 = (long)K1 * G1;
  const long total2 = (long)K2 * G2;
  const long total  = total1 + total2 + G1 + G2;
  long idx    = (long)blockIdx.x * blockDim.x + threadIdx.x;
  long stride = (long)gridDim.x * blockDim.x;
  for (long i = idx; i < total; i += stride) {
    if (i < total1) {
      int k = (int)(i / G1), n = (int)(i % G1);
      float v = (k < F_SZ) ? Wih1[(long)n * F_SZ + k] : Whh1[(long)n * H_SZ + (k - F_SZ)];
      int lane, slot;
      frag_pos(k & 31, n & 15, lane, slot);
      w1p[(((long)(n >> 4) * KT1 + (k >> 5)) * 32 + lane) * 16 + slot] = f2bf(v);
    } else if (i < total1 + total2) {
      long j = i - total1;
      int k = (int)(j / G2), n = (int)(j % G2);
      float v = (k < H_SZ) ? Wih2[(long)n * H_SZ + k] : Whh2[(long)n * F_SZ + (k - H_SZ)];
      int lane, slot;
      frag_pos(k & 31, n & 15, lane, slot);
      w2p[(((long)(n >> 4) * KT2 + (k >> 5)) * 32 + lane) * 16 + slot] = f2bf(v);
    } else if (i < total1 + total2 + G1) {
      int n = (int)(i - total1 - total2);
      b1[n] = bih1[n] + bhh1[n];
    } else {
      int n = (int)(i - total1 - total2 - G1);
      b2[n] = bih2[n] + bhh2[n];
    }
  }
}

// B-fragment addresses in the packed streams (consumption order: kt outer)
__device__ __forceinline__ const bf16* b1addr(const bf16* w1p, int w, int lane, int lin) {
  int kt = lin >> 3, idx = lin & 7;
  int ntile = (idx >> 1) * 32 + w * 2 + (idx & 1);
  return w1p + ((long)ntile * KT1 + kt) * 512 + lane * 16;
}
__device__ __forceinline__ const bf16* b2addr(const bf16* w2p, int w, int lane, int lin) {
  int kt = lin >> 2, g = lin & 3;
  int ntile = g * 8 + w;
  return w2p + ((long)ntile * KT2 + kt) * 512 + lane * 16;
}

// A-fragment gather (per-lane two 16B chunks, ISA 16-bit A layout)
__device__ __forceinline__ v16bf gatherA(const bf16* rowp) {
  v8bf alo = *(const v8bf*)(rowp);
  v8bf ahi = *(const v8bf*)(rowp + 16);
  v16bf a;
  #pragma unroll
  for (int q = 0; q < 8; ++q) { a[q] = alo[q]; a[q + 8] = ahi[q]; }
  return a;
}

// ---------------------------------------------------------------------------
// Main scan kernel.
//  gates1: all 16 waves; wave w owns h1 cols [w*32, w*32+32) in every gate
//          block -> acc1[4][2], c1 cell state lives in VGPRs.
//  gates2: waves 0-7 own f cols [w*16, w*16+16) -> acc2[4], d state in VGPRs;
//          waves 8-15 concurrently async-stage x_{t+1} into parity-flipped LDS.
// ---------------------------------------------------------------------------
__global__ __launch_bounds__(512)
void lstm_scan_kernel(const float* __restrict__ data,
                      const bf16*  __restrict__ w1p,
                      const bf16*  __restrict__ w2p,
                      const float* __restrict__ b1,
                      const float* __restrict__ b2,
                      float* __restrict__ out) {
  __shared__ __align__(16) bf16  sH1 [MT * LDH];        // h1 bf16 (K = 128..639 of A1)
  __shared__ __align__(16) bf16  sXbf[2][MT * LDXB];    // x_t bf16, double-buffered
  __shared__ __align__(16) bf16  sA2 [MT * LDA2];       // [c1 bf16 | h2 bf16]
  __shared__ __align__(16) float sXf [2][MT * LDX];     // x_t f32 (residual), dbl-buf

  const int tid  = threadIdx.x;
  const int lane = tid & 31;
  const int w    = tid >> 5;     // wave id 0..15
  const int wg   = blockIdx.x;   // batch rows [wg*16, wg*16+16)
  const int nn   = lane & 15;
  const int hi   = lane >> 4;
  const int tloc = tid & 255;    // staging-thread index within waves 8..15

  BfBits z; z.u = 0;
  for (int i = tid; i < MT * LDH;  i += 512) sH1[i] = z.b;
  for (int i = tid; i < MT * LDA2; i += 512) sA2[i] = z.b;
  float c1s[2][8];
  #pragma unroll
  for (int j = 0; j < 2; ++j)
    #pragma unroll
    for (int r = 0; r < 8; ++r) c1s[j][r] = 0.0f;
  float dcs[8];
  #pragma unroll
  for (int r = 0; r < 8; ++r) dcs[r] = 0.0f;

  // ---- initial stage of x_0 (waves 8..15), async global -> LDS ----------
  if (w >= 8) {
    #pragma unroll
    for (int it = 0; it < 2; ++it) {
      int chunk = tloc + it * 256;          // 512 chunks * 16B = 16x128 f32
      int d0 = chunk * 4, row = d0 >> 7, col = d0 & 127;
      unsigned lds = (unsigned)(unsigned long long)(&sXf[0][row * LDX + col]);
      unsigned long long ga = (unsigned long long)(const void*)
          (data + (long)(wg * MT + row) * (T_SZ * F_SZ) + col);
      asm volatile("global_load_async_to_lds_b128 %0, %1, off"
                   :: "v"(lds), "v"(ga) : "memory");
    }
    asm volatile("s_wait_asynccnt 0" ::: "memory");
    #pragma unroll
    for (int it = 0; it < 2; ++it) {
      int chunk = tloc + it * 256;
      int d0 = chunk * 4, row = d0 >> 7, col = d0 & 127;
      #pragma unroll
      for (int q = 0; q < 4; ++q)
        sXbf[0][row * LDXB + col + q] = f2bf(sXf[0][row * LDX + col + q]);
    }
  }
  __syncthreads();

  for (int t = 0; t < T_SZ; ++t) {
    const int p = t & 1;
    const bf16* xb = &sXbf[p][0];

    // ---- gates1 = [x|h1] @ B1 + bias --------------------------------------
    v8f acc1[4][2];
    #pragma unroll
    for (int g = 0; g < 4; ++g)
      #pragma unroll
      for (int j = 0; j < 2; ++j) {
        float bv = b1[g * 512 + w * 32 + j * 16 + nn];
        v8f a;
        #pragma unroll
        for (int r = 0; r < 8; ++r) a[r] = bv;
        acc1[g][j] = a;
      }
    // 3-deep B pipeline + A one K-tile ahead
    v16bf bq0 = *(const v16bf*)b1addr(w1p, w, lane, 0);
    v16bf bq1 = *(const v16bf*)b1addr(w1p, w, lane, 1);
    v16bf bq2 = *(const v16bf*)b1addr(w1p, w, lane, 2);
    v16bf a_cur = gatherA(xb + nn * LDXB + hi * 8);   // kt = 0 is in x region
    #pragma unroll 1
    for (int kt = 0; kt < KT1; ++kt) {
      v16bf a_nxt = a_cur;
      if (kt + 1 < KT1) {
        int koff = (kt + 1) * 32 + hi * 8;
        const bf16* rowp = (kt + 1 < 4) ? (xb + nn * LDXB + koff)
                                        : (sH1 + nn * LDH + (koff - F_SZ));
        a_nxt = gatherA(rowp);
      }
      __builtin_prefetch(b1addr(w1p, w, lane, (kt + 2 < KT1 ? kt + 2 : kt) * 8), 0, 1);
      #pragma unroll
      for (int idx = 0; idx < 8; ++idx) {
        int lin = kt * 8 + idx + 3;
        v16bf bnew = bq0;
        if (lin < KT1 * 8) bnew = *(const v16bf*)b1addr(w1p, w, lane, lin);
        acc1[idx >> 1][idx & 1] = __builtin_amdgcn_wmma_f32_16x16x32_bf16(
            false, a_cur, false, bq0, (short)0, acc1[idx >> 1][idx & 1], false, false);
        bq0 = bq1; bq1 = bq2; bq2 = bnew;
      }
      a_cur = a_nxt;
    }
    __syncthreads();  // everyone done reading old h1 / x_t fragments

    // ---- lstm1 activations (c1 state in registers) ----------------------
    #pragma unroll
    for (int j = 0; j < 2; ++j) {
      int hcol = w * 32 + j * 16 + nn;
      #pragma unroll
      for (int r = 0; r < 8; ++r) {
        int m = r + 8 * hi;
        float ig = sigm(acc1[0][j][r]);
        float fg = sigm(acc1[1][j][r]);
        float gg = tanh_fast(acc1[2][j][r]);
        float og = sigm(acc1[3][j][r]);
        float cn = fg * c1s[j][r] + ig * gg;
        float hn = og * tanh_fast(cn);
        c1s[j][r] = cn;
        sH1[m * LDH + hcol]  = f2bf(hn);  // h1 -> next step's A1
        sA2[m * LDA2 + hcol] = f2bf(cn);  // c1 -> this step's A2
      }
    }
    __syncthreads();

    if (w < 8) {
      // ---- gates2 = [c1|h2] @ B2 + bias ; 3-deep pipelined --------------
      v8f acc2[4];
      #pragma unroll
      for (int g = 0; g < 4; ++g) {
        float bv = b2[g * 128 + w * 16 + nn];
        v8f a;
        #pragma unroll
        for (int r = 0; r < 8; ++r) a[r] = bv;
        acc2[g] = a;
      }
      v16bf cq0 = *(const v16bf*)b2addr(w2p, w, lane, 0);
      v16bf cq1 = *(const v16bf*)b2addr(w2p, w, lane, 1);
      v16bf cq2 = *(const v16bf*)b2addr(w2p, w, lane, 2);
      v16bf a2_cur = gatherA(sA2 + nn * LDA2 + hi * 8);
      #pragma unroll 1
      for (int kt = 0; kt < KT2; ++kt) {
        v16bf a2_nxt = a2_cur;
        if (kt + 1 < KT2)
          a2_nxt = gatherA(sA2 + nn * LDA2 + (kt + 1) * 32 + hi * 8);
        #pragma unroll
        for (int g = 0; g < 4; ++g) {
          int lin = kt * 4 + g + 3;
          v16bf bnew = cq0;
          if (lin < KT2 * 4) bnew = *(const v16bf*)b2addr(w2p, w, lane, lin);
          acc2[g] = __builtin_amdgcn_wmma_f32_16x16x32_bf16(
              false, a2_cur, false, cq0, (short)0, acc2[g], false, false);
          cq0 = cq1; cq1 = cq2; cq2 = bnew;
        }
        a2_cur = a2_nxt;
      }
      __syncthreads();  // done reading c1/h2 before h2 overwrite; x_{t+1} staged

      // ---- lstm2 activations + residual output (d state in registers) ---
      int fcol = w * 16 + nn;
      #pragma unroll
      for (int r = 0; r < 8; ++r) {
        int m = r + 8 * hi;
        float ig = sigm(acc2[0][r]);
        float fg = sigm(acc2[1][r]);
        float gg = tanh_fast(acc2[2][r]);
        float og = sigm(acc2[3][r]);
        float dn = fg * dcs[r] + ig * gg;
        float h2 = og * tanh_fast(dn);
        dcs[r] = dn;
        sA2[m * LDA2 + H_SZ + fcol] = f2bf(h2);  // h2 -> next step's A2
        out[((long)(wg * MT + m) * T_SZ + t) * F_SZ + fcol] = dn + sXf[p][m * LDX + fcol];
      }
    } else {
      // ---- waves 8..15: async-stage x_{t+1} into parity-flipped buffers --
      if (t + 1 < T_SZ) {
        const int np = p ^ 1;
        #pragma unroll
        for (int it = 0; it < 2; ++it) {
          int chunk = tloc + it * 256;
          int d0 = chunk * 4, row = d0 >> 7, col = d0 & 127;
          unsigned lds = (unsigned)(unsigned long long)(&sXf[np][row * LDX + col]);
          unsigned long long ga = (unsigned long long)(const void*)
              (data + (long)(wg * MT + row) * (T_SZ * F_SZ) + (long)(t + 1) * F_SZ + col);
          asm volatile("global_load_async_to_lds_b128 %0, %1, off"
                       :: "v"(lds), "v"(ga) : "memory");
        }
        asm volatile("s_wait_asynccnt 0" ::: "memory");
        #pragma unroll
        for (int it = 0; it < 2; ++it) {
          int chunk = tloc + it * 256;
          int d0 = chunk * 4, row = d0 >> 7, col = d0 & 127;
          #pragma unroll
          for (int q = 0; q < 4; ++q)
            sXbf[np][row * LDXB + col + q] = f2bf(sXf[np][row * LDX + col + q]);
        }
      }
      __syncthreads();  // pairs with waves 0-7 barrier above
    }
    __syncthreads();  // h2/out written; staging visible to all before next t
  }
}

// ---------------------------------------------------------------------------
extern "C" void kernel_launch(void* const* d_in, const int* in_sizes, int n_in,
                              void* d_out, int out_size, void* d_ws, size_t ws_size,
                              hipStream_t stream) {
  const float* data = (const float*)d_in[0];
  const float* Wih1 = (const float*)d_in[1];
  const float* Whh1 = (const float*)d_in[2];
  const float* bih1 = (const float*)d_in[3];
  const float* bhh1 = (const float*)d_in[4];
  const float* Wih2 = (const float*)d_in[5];
  const float* Whh2 = (const float*)d_in[6];
  const float* bih2 = (const float*)d_in[7];
  const float* bhh2 = (const float*)d_in[8];
  float* out = (float*)d_out;

  // workspace layout: W1 pack (2.62 MB bf16) | W2 pack (0.66 MB) | b1 | b2
  bf16*  w1p = (bf16*)d_ws;
  bf16*  w2p = w1p + (size_t)NT1 * KT1 * 512;
  float* b1  = (float*)(w2p + (size_t)NT2 * KT2 * 512);
  float* b2  = b1 + G1;

  lstm_pack_kernel<<<2048, 256, 0, stream>>>(Wih1, Whh1, bih1, bhh1,
                                             Wih2, Whh2, bih2, bhh2,
                                             w1p, w2p, b1, b2);
  lstm_scan_kernel<<<B_SZ / MT, 512, 0, stream>>>(data, w1p, w2p, b1, b2, out);
}
